// Loss_29592324669863
// MI455X (gfx1250) — matrix operand order repaired and checked by reading, not compiled, not run on hardware
//
#include <hip/hip_runtime.h>
#include <hip/hip_bf16.h>

// ---------------------------------------------------------------------------
// MI455X (gfx1250, wave32) fused contrastive-loss kernel.
// Dominant cost: three Gram matrices (A·A^T style), ~51.5 GFLOP total.
// Memory traffic is tiny if sims are never materialized -> compute bound ->
// route the matmul through V_WMMA_F32_16X16X32_F16 (f32 accumulate).
// ---------------------------------------------------------------------------

typedef _Float16 v16h  __attribute__((ext_vector_type(16)));
typedef _Float16 half8 __attribute__((ext_vector_type(8)));
typedef float    v8f   __attribute__((ext_vector_type(8)));

#define FEAT_D 256  // feature dim for every matrix in this problem

// A-matrix fragment (16x32 f16, MxK). ISA 7.12.2 layout:
//   lanes 0-15  (row M=lane):    elems 0-7 = K kb+0..7,  elems 8-15 = K kb+16..23
//   lanes 16-31 (row M=lane-16): elems 0-7 = K kb+8..15, elems 8-15 = K kb+24..31
__device__ __forceinline__ v16h load_fragA(const _Float16* __restrict__ base,
                                           int rowBase, int kb, int lane) {
    int h = lane >> 4;
    int r = rowBase + (lane & 15);
    const _Float16* p = base + r * FEAT_D + kb + h * 8;
    half8 lo = *(const half8*)(p);        // 16B aligned
    half8 hi = *(const half8*)(p + 16);   // 16B aligned
    v16h out;
#pragma unroll
    for (int i = 0; i < 8; ++i) { out[i] = lo[i]; out[i + 8] = hi[i]; }
    return out;
}

// B-matrix fragment (32x16 f16, KxN), column n = stored row (colBase+n).
// Layout (matches A): lanes 0-15 hold K kb+0..7 and kb+16..23,
// lanes 16-31 hold K kb+8..15 and kb+24..31. Two 16B reads.
__device__ __forceinline__ v16h load_fragB(const _Float16* __restrict__ base,
                                           int colBase, int kb, int lane) {
    int c = colBase + (lane & 15);
    const _Float16* p = base + c * FEAT_D + kb + (lane >> 4) * 8;
    half8 lo = *(const half8*)(p);
    half8 hi = *(const half8*)(p + 16);
    v16h out;
#pragma unroll
    for (int i = 0; i < 8; ++i) { out[i] = lo[i]; out[i + 8] = hi[i]; }
    return out;
}

// Row-normalize f32 -> f16 (cosine prep). One block per row, 256 threads.
__global__ void normalize_rows(const float* __restrict__ in,
                               _Float16* __restrict__ out) {
    __shared__ float red[256];
    int row = blockIdx.x;
    int t = threadIdx.x;
    float v = in[row * FEAT_D + t];
    red[t] = v * v;
    __syncthreads();
#pragma unroll
    for (int s = 128; s > 0; s >>= 1) {
        if (t < s) red[t] += red[t + s];
        __syncthreads();
    }
    float nrm = fmaxf(sqrtf(red[0]), 1e-8f);
    out[row * FEAT_D + t] = (_Float16)(v / nrm);
}

// MODE 0: contrastive with clamp(exp,1e10)  -> num/den per row
// MODE 1: contrastive, no clamp             -> num/den per row
// MODE 2: hash loss                         -> single scalar accumulator
template <int MODE>
__global__ void __launch_bounds__(32)
simReduceKernel(const _Float16* __restrict__ A,
                const _Float16* __restrict__ B,
                const int* __restrict__ labels,
                const float* __restrict__ tempPtr,
                const float* __restrict__ xiPtr,
                float* __restrict__ numOut,
                float* __restrict__ denOut,
                float* __restrict__ hashOut,
                int colTilesPerBlock, int labMask) {
    const int lane = threadIdx.x;                 // wave32: one wave per block
    const int rowBase = blockIdx.x * 16;
    const int colTile0 = blockIdx.y * colTilesPerBlock;

    const float invT = 1.0f / tempPtr[0];
    const float xi = (MODE == 2) ? xiPtr[0] : 0.0f;

    // Keep the whole 16x256 A panel resident in VGPRs (8 frags = 64 VGPRs).
    v16h afrag[8];
#pragma unroll
    for (int k = 0; k < 8; ++k) afrag[k] = load_fragA(A, rowBase, k * 32, lane);

    // C/D layout: lane L, acc[r] = sim[rowBase + r + 8*(L>>4)][colBase + (L&15)]
    int rowLab[8];
#pragma unroll
    for (int r = 0; r < 8; ++r)
        rowLab[r] = labels[(rowBase + r + 8 * (lane >> 4)) & labMask];

    float accNum[8] = {0, 0, 0, 0, 0, 0, 0, 0};
    float accDen[8] = {0, 0, 0, 0, 0, 0, 0, 0};
    float hashAcc = 0.0f;

    for (int ct = 0; ct < colTilesPerBlock; ++ct) {
        int colBase = (colTile0 + ct) * 16;
        v8f c = {};
#pragma unroll
        for (int k = 0; k < 8; ++k) {
            v16h bfrag = load_fragB(B, colBase, k * 32, lane);
            c = __builtin_amdgcn_wmma_f32_16x16x32_f16(
                    false, afrag[k], false, bfrag, (short)0, c, false, false);
        }
        int colLab = labels[(colBase + (lane & 15)) & labMask];
#pragma unroll
        for (int r = 0; r < 8; ++r) {
            float s = c[r];
            if (MODE == 2) {
                float Dm = (1.0f - s) * 0.5f;
                if (rowLab[r] == colLab) {
                    hashAcc += Dm * Dm;
                } else {
                    float d = fmaxf(xi - Dm, 0.0f);
                    hashAcc += d * d;
                }
            } else {
                float e = __expf(s * invT);
                if (MODE == 0) e = fminf(e, 1e10f);
                accDen[r] += e;
                if (rowLab[r] == colLab) accNum[r] += e;
            }
        }
    }

    if (MODE == 2) {
#pragma unroll
        for (int off = 16; off > 0; off >>= 1)
            hashAcc += __shfl_xor(hashAcc, off, 32);
        if (lane == 0) atomicAdd(hashOut, hashAcc);
    } else {
        // Each 16-lane half-wave owns one set of 8 rows: reduce across lanes.
#pragma unroll
        for (int r = 0; r < 8; ++r) {
            float n = accNum[r], d = accDen[r];
#pragma unroll
            for (int off = 8; off > 0; off >>= 1) {
                n += __shfl_xor(n, off, 16);
                d += __shfl_xor(d, off, 16);
            }
            if ((lane & 15) == 0) {
                int row = rowBase + r + 8 * (lane >> 4);
                atomicAdd(&numOut[row], n);
                atomicAdd(&denOut[row], d);
            }
        }
    }
}

__global__ void finalizeKernel(const float* __restrict__ num1,
                               const float* __restrict__ den1,
                               const float* __restrict__ num2,
                               const float* __restrict__ den2,
                               const float* __restrict__ hashSum,
                               float* __restrict__ out,
                               int n1, int n2, float hashScale) {
    __shared__ float red1[256];
    __shared__ float red2[256];
    int t = threadIdx.x;
    float a1 = 0.0f, a2 = 0.0f;
    for (int i = t; i < n1; i += 256) {
        float d = fmaxf(den1[i], 1e-10f);
        a1 -= __logf(num1[i] / d);
    }
    for (int i = t; i < n2; i += 256) {
        float d = fmaxf(den2[i], 1e-10f);
        a2 -= __logf(num2[i] / d);
    }
    red1[t] = a1;
    red2[t] = a2;
    __syncthreads();
#pragma unroll
    for (int s = 128; s > 0; s >>= 1) {
        if (t < s) { red1[t] += red1[t + s]; red2[t] += red2[t + s]; }
        __syncthreads();
    }
    if (t == 0) {
        out[0] = red1[0] / (float)n1;
        out[1] = red2[0] / (float)n2;
        out[2] = hashSum[0] * hashScale;
    }
}

extern "C" void kernel_launch(void* const* d_in, const int* in_sizes, int n_in,
                              void* d_out, int out_size, void* d_ws, size_t ws_size,
                              hipStream_t stream) {
    (void)in_sizes; (void)n_in; (void)out_size; (void)ws_size;

    const int N = 4096, M = 2, MN = M * N;  // D = 256

    const float* label_enc = (const float*)d_in[0];   // [N, 256]
    const float* fusion_z  = (const float*)d_in[1];   // [M, N, 256]
    const float* samp_emb  = (const float*)d_in[2];   // [N, 256]
    const float* lab_emb   = (const float*)d_in[3];   // [N, 256]
    const int*   labels    = (const int*)d_in[4];     // [N]
    const float* temp      = (const float*)d_in[5];   // scalar
    const float* xi        = (const float*)d_in[6];   // scalar
    float* out = (float*)d_out;

    // Workspace carve-out (~10.6 MB)
    char* ws = (char*)d_ws;
    _Float16* An = (_Float16*)ws;  ws += (size_t)N  * FEAT_D * sizeof(_Float16);
    _Float16* Fn = (_Float16*)ws;  ws += (size_t)MN * FEAT_D * sizeof(_Float16);
    _Float16* Sn = (_Float16*)ws;  ws += (size_t)N  * FEAT_D * sizeof(_Float16);
    _Float16* Ln = (_Float16*)ws;  ws += (size_t)N  * FEAT_D * sizeof(_Float16);
    float* num1 = (float*)ws;      ws += (size_t)N  * sizeof(float);
    float* den1 = (float*)ws;      ws += (size_t)N  * sizeof(float);
    float* num2 = (float*)ws;      ws += (size_t)MN * sizeof(float);
    float* den2 = (float*)ws;      ws += (size_t)MN * sizeof(float);
    float* hashSum = (float*)ws;   ws += sizeof(float);

    // Normalize rows -> f16 panels
    normalize_rows<<<N,  256, 0, stream>>>(label_enc, An);
    normalize_rows<<<MN, 256, 0, stream>>>(fusion_z,  Fn);
    normalize_rows<<<N,  256, 0, stream>>>(samp_emb,  Sn);
    normalize_rows<<<N,  256, 0, stream>>>(lab_emb,   Ln);

    // Zero accumulators (num1..hashSum are contiguous)
    hipMemsetAsync(num1, 0, (size_t)(2 * N + 2 * MN + 1) * sizeof(float), stream);

    const int labMask = N - 1;  // loss2 tiles labels: index % N (N power of 2)
    const int COLSPLIT = 8;

    // loss_class: An x An^T, 256 row-tiles x (256/8=32 col-tiles per block)
    simReduceKernel<0><<<dim3(N / 16, COLSPLIT), 32, 0, stream>>>(
        An, An, labels, temp, xi, num1, den1, nullptr, (N / 16) / COLSPLIT, labMask);

    // loss_cross: Fn x Fn^T, 512 row-tiles x 64 col-tiles per block
    simReduceKernel<1><<<dim3(MN / 16, COLSPLIT), 32, 0, stream>>>(
        Fn, Fn, labels, temp, xi, num2, den2, nullptr, (MN / 16) / COLSPLIT, labMask);

    // loss_hash: Sn x Ln^T
    simReduceKernel<2><<<dim3(N / 16, COLSPLIT), 32, 0, stream>>>(
        Sn, Ln, labels, temp, xi, nullptr, nullptr, hashSum, (N / 16) / COLSPLIT, labMask);

    finalizeKernel<<<1, 256, 0, stream>>>(num1, den1, num2, den2, hashSum, out,
                                          N, MN, 1.0f / ((float)N * (float)N));
}
